// SimpleHeteroGAT_33784212750818
// MI455X (gfx1250) — matrix operand order repaired and checked by reading, not compiled, not run on hardware
//
#include <hip/hip_runtime.h>
#include <hip/hip_bf16.h>
#include <cstddef>

typedef __attribute__((ext_vector_type(16))) __bf16 v16bf;
typedef __attribute__((ext_vector_type(8)))  __bf16 v8bf;
typedef __attribute__((ext_vector_type(8)))  float  v8f;

union B16 { v16bf v; v8bf h[2]; };

#define LDS_STRIDE 136   // 128 + 8 bf16 pad -> bank-optimal ds_load_b128

// ---------------------------------------------------------------------------
// Zero-fill helper (grid-stride)
// ---------------------------------------------------------------------------
__global__ void zero_f32(float* __restrict__ p, size_t n) {
  size_t i = (size_t)blockIdx.x * blockDim.x + threadIdx.x;
  size_t stride = (size_t)gridDim.x * blockDim.x;
  for (; i < n; i += stride) p[i] = 0.0f;
}

// ---------------------------------------------------------------------------
// Projection: h = x @ W_proj + b_proj for user rows [0,nu) and item rows.
// W_proj is staged once per block into LDS as bf16, column-major + pad, so
// each B tile is two ds_load_b128 instead of 16 strided global loads.
// One wave computes 16 rows x 128 cols per tile; waves grid-stride over tiles.
// ---------------------------------------------------------------------------
__global__ void __launch_bounds__(256) proj_wmma(
    const float* __restrict__ xu, const float* __restrict__ xi,
    const float* __restrict__ W,  const float* __restrict__ b,
    float* __restrict__ hu, float* __restrict__ hi, int nu, int ni)
{
  __shared__ __attribute__((aligned(16))) __bf16 Wl[128 * LDS_STRIDE];

  // ---- cooperative stage: W (128x128 f32 row-major) -> LDS bf16 col-major
  for (int i = threadIdx.x; i < (128 * 128) / 4; i += 256) {
    int L = i << 2;
    int k = L >> 7;          // K row
    int n = L & 127;         // N col base
    float4 w = ((const float4*)W)[i];
    Wl[(n + 0) * LDS_STRIDE + k] = (__bf16)w.x;
    Wl[(n + 1) * LDS_STRIDE + k] = (__bf16)w.y;
    Wl[(n + 2) * LDS_STRIDE + k] = (__bf16)w.z;
    Wl[(n + 3) * LDS_STRIDE + k] = (__bf16)w.w;
  }
  __syncthreads();

  int lane = threadIdx.x & 31;
  int m  = lane & 15;                          // row/col within tile
  int kh = lane >> 4;                          // half-wave selector
  int ntiles = (nu + ni) >> 4;
  int wave   = (int)(blockIdx.x * 8 + (threadIdx.x >> 5));
  int nwaves = (int)(gridDim.x * 8);

  for (int wt = wave; wt < ntiles; wt += nwaves) {   // wave-uniform loop
    int rb = wt << 4;
    const float* X; float* H;
    if (rb < nu) { X = xu + (size_t)rb * 128; H = hu + (size_t)rb * 128; }
    else         { X = xi + (size_t)(rb - nu) * 128; H = hi + (size_t)(rb - nu) * 128; }

    // A tiles (16x32 bf16): elems 0..7 = K kt*32+kh*8+{0..7}; 8..15 = +16
    const float* xrow = X + (size_t)m * 128;
    v16bf A[4];
#pragma unroll
    for (int kt = 0; kt < 4; ++kt) {
      const float4* p = (const float4*)(xrow + kt * 32 + kh * 8);
      float4 a0 = p[0], a1 = p[1];     // K +0..7
      float4 a2 = p[4], a3 = p[5];     // K +16..23
      A[kt][0] = (__bf16)a0.x; A[kt][1] = (__bf16)a0.y;
      A[kt][2] = (__bf16)a0.z; A[kt][3] = (__bf16)a0.w;
      A[kt][4] = (__bf16)a1.x; A[kt][5] = (__bf16)a1.y;
      A[kt][6] = (__bf16)a1.z; A[kt][7] = (__bf16)a1.w;
      A[kt][8]  = (__bf16)a2.x; A[kt][9]  = (__bf16)a2.y;
      A[kt][10] = (__bf16)a2.z; A[kt][11] = (__bf16)a2.w;
      A[kt][12] = (__bf16)a3.x; A[kt][13] = (__bf16)a3.y;
      A[kt][14] = (__bf16)a3.z; A[kt][15] = (__bf16)a3.w;
    }

#pragma unroll
    for (int nt = 0; nt < 8; ++nt) {
      int nb = nt * 16;
      float bv = b[nb + m];
      v8f c = {bv, bv, bv, bv, bv, bv, bv, bv};
#pragma unroll
      for (int kt = 0; kt < 4; ++kt) {
        // B tile 32x16 from LDS: lane = column nb+m, K = kt*32+kh*8+{0..7}, +16+{0..7}
        B16 bu;
        const v8bf* bp =
            (const v8bf*)&Wl[(size_t)(nb + m) * LDS_STRIDE + kt * 32 + kh * 8];
        bu.h[0] = bp[0];
        bu.h[1] = bp[2];
        c = __builtin_amdgcn_wmma_f32_16x16x32_bf16(false, A[kt], false, bu.v,
                                                    (short)0, c, false, false);
      }
#pragma unroll
      for (int r = 0; r < 8; ++r)           // C: VGPR r -> row (kh?8:0)+r, col m
        H[(size_t)((kh << 3) + r) * 128 + nb + m] = c[r];
    }
  }
}

// ---------------------------------------------------------------------------
// Edge scatter: acc[dst] += al * h[src]; deg[dst] += 1.
// (The ar*h[dst] term is folded in later via the degree count.)
// ---------------------------------------------------------------------------
__global__ void __launch_bounds__(256) edge_scatter(
    const int* __restrict__ src, const int* __restrict__ dst,
    const float* __restrict__ h, const float* __restrict__ alp,
    float* __restrict__ acc, float* __restrict__ deg, int ne)
{
  int e = (int)((blockIdx.x * (size_t)blockDim.x + threadIdx.x) >> 5);
  if (e >= ne) return;
  int lane = threadIdx.x & 31;
  float al = alp[0];
  int s = src[e], d = dst[e];
  float4 v = ((const float4*)(h + (size_t)s * 128))[lane];
  float* a = acc + (size_t)d * 128 + lane * 4;
  atomicAdd(a + 0, al * v.x);
  atomicAdd(a + 1, al * v.y);
  atomicAdd(a + 2, al * v.z);
  atomicAdd(a + 3, al * v.w);
  if (lane == 0) atomicAdd(&deg[d], 1.0f);
}

// ---------------------------------------------------------------------------
// Output: out_user = elu(acc + (ar1*deg1 + ar2*deg2)*h_user) @ W_out + b_out.
// W_out staged to LDS bf16 col-major; degree-combine + ELU fused into A load.
// ---------------------------------------------------------------------------
__global__ void __launch_bounds__(256) out_wmma(
    const float* __restrict__ acc, const float* __restrict__ hu,
    const float* __restrict__ deg1, const float* __restrict__ deg2,
    const float* __restrict__ ar1p, const float* __restrict__ ar2p,
    const float* __restrict__ Wo, const float* __restrict__ bo,
    float* __restrict__ out, int nu)
{
  __shared__ __attribute__((aligned(16))) __bf16 Wl[64 * LDS_STRIDE];

  // ---- stage W_out (128x64 f32 row-major) -> LDS bf16 col-major
  for (int i = threadIdx.x; i < (128 * 64) / 4; i += 256) {
    int L = i << 2;
    int k = L >> 6;          // K row
    int n = L & 63;          // N col base
    float4 w = ((const float4*)Wo)[i];
    Wl[(n + 0) * LDS_STRIDE + k] = (__bf16)w.x;
    Wl[(n + 1) * LDS_STRIDE + k] = (__bf16)w.y;
    Wl[(n + 2) * LDS_STRIDE + k] = (__bf16)w.z;
    Wl[(n + 3) * LDS_STRIDE + k] = (__bf16)w.w;
  }
  __syncthreads();

  int lane = threadIdx.x & 31;
  int m  = lane & 15;
  int kh = lane >> 4;
  float r1 = ar1p[0], r2 = ar2p[0];
  int ntiles = nu >> 4;
  int wave   = (int)(blockIdx.x * 8 + (threadIdx.x >> 5));
  int nwaves = (int)(gridDim.x * 8);

  for (int wt = wave; wt < ntiles; wt += nwaves) {   // wave-uniform loop
    int rb = wt << 4;
    int row = rb + m;
    float coeff = r1 * deg1[row] + r2 * deg2[row];
    const float* arow = acc + (size_t)row * 128;
    const float* hrow = hu  + (size_t)row * 128;

    v16bf A[4];
#pragma unroll
    for (int kt = 0; kt < 4; ++kt) {
      int kb = kt * 32 + kh * 8;
      const float4* pa = (const float4*)(arow + kb);
      const float4* ph = (const float4*)(hrow + kb);
      float4 a0 = pa[0], a1 = pa[1], a2 = pa[4], a3 = pa[5];
      float4 h0 = ph[0], h1 = ph[1], h2 = ph[4], h3 = ph[5];
      float x[16] = {
        a0.x + coeff * h0.x, a0.y + coeff * h0.y, a0.z + coeff * h0.z, a0.w + coeff * h0.w,
        a1.x + coeff * h1.x, a1.y + coeff * h1.y, a1.z + coeff * h1.z, a1.w + coeff * h1.w,
        a2.x + coeff * h2.x, a2.y + coeff * h2.y, a2.z + coeff * h2.z, a2.w + coeff * h2.w,
        a3.x + coeff * h3.x, a3.y + coeff * h3.y, a3.z + coeff * h3.z, a3.w + coeff * h3.w };
#pragma unroll
      for (int e = 0; e < 16; ++e) {
        float v = x[e];
        v = v > 0.0f ? v : (expf(v) - 1.0f);   // ELU
        A[kt][e] = (__bf16)v;
      }
    }

#pragma unroll
    for (int nt = 0; nt < 4; ++nt) {          // OUT = 64 -> 4 column tiles
      int nb = nt * 16;
      float bv = bo[nb + m];
      v8f c = {bv, bv, bv, bv, bv, bv, bv, bv};
#pragma unroll
      for (int kt = 0; kt < 4; ++kt) {
        B16 bu;
        const v8bf* bp =
            (const v8bf*)&Wl[(size_t)(nb + m) * LDS_STRIDE + kt * 32 + kh * 8];
        bu.h[0] = bp[0];
        bu.h[1] = bp[2];
        c = __builtin_amdgcn_wmma_f32_16x16x32_bf16(false, A[kt], false, bu.v,
                                                    (short)0, c, false, false);
      }
#pragma unroll
      for (int r = 0; r < 8; ++r)
        out[(size_t)(rb + (kh << 3) + r) * 64 + nb + m] = c[r];
    }
  }
}

// ---------------------------------------------------------------------------
extern "C" void kernel_launch(void* const* d_in, const int* in_sizes, int n_in,
                              void* d_out, int out_size, void* d_ws, size_t ws_size,
                              hipStream_t stream) {
  (void)n_in; (void)out_size; (void)ws_size;
  const float* x_user = (const float*)d_in[0];
  const float* x_item = (const float*)d_in[1];
  const float* W_proj = (const float*)d_in[2];
  const float* b_proj = (const float*)d_in[3];
  const float* W_out  = (const float*)d_in[4];
  const float* b_out  = (const float*)d_in[5];
  const float* al1 = (const float*)d_in[6];
  const float* ar1 = (const float*)d_in[7];
  const float* al2 = (const float*)d_in[8];
  const float* ar2 = (const float*)d_in[9];
  const int* e1_src = (const int*)d_in[10];
  const int* e1_dst = (const int*)d_in[11];
  const int* e2_src = (const int*)d_in[12];
  const int* e2_dst = (const int*)d_in[13];

  const int nu = in_sizes[0] / 128;
  const int ni = in_sizes[1] / 128;
  const int e1 = in_sizes[10];
  const int e2 = in_sizes[12];

  float* out = (float*)d_out;                  // [nu*64] out_user ++ [ni*128] out_item
  float* acc = out + (size_t)nu * 64;          // reuse out_item region as accumulator
  float* hu   = (float*)d_ws;                  // [nu*128]
  float* hi   = hu + (size_t)nu * 128;         // [ni*128]
  float* deg1 = hi + (size_t)ni * 128;         // [nu]
  float* deg2 = deg1 + nu;                     // [nu]

  // 1) zero accumulator + degree counters
  zero_f32<<<2048, 256, 0, stream>>>(acc, (size_t)nu * 128);
  zero_f32<<<256, 256, 0, stream>>>(deg1, (size_t)2 * nu);

  // 2) shared projection (WMMA bf16, f32 accumulate; W in LDS)
  proj_wmma<<<512, 256, 0, stream>>>(x_user, x_item, W_proj, b_proj,
                                     hu, hi, nu, ni);

  // 3) edge scatter (al*h_src only; ar*h_dst folded via degree)
  edge_scatter<<<(e1 + 7) / 8, 256, 0, stream>>>(e1_src, e1_dst, hu, al1, acc, deg1, e1);
  edge_scatter<<<(e2 + 7) / 8, 256, 0, stream>>>(e2_src, e2_dst, hi, al2, acc, deg2, e2);

  // 4) fused degree-combine + ELU + output GEMM (WMMA; W_out in LDS)
  out_wmma<<<512, 256, 0, stream>>>(acc, hu, deg1, deg2, ar1, ar2,
                                    W_out, b_out, out, nu);

  // 5) out_item = elu(0) = 0 (accumulator region becomes the final output)
  zero_f32<<<2048, 256, 0, stream>>>(acc, (size_t)ni * 128);
}
